// CausalRoPEAttention_9766755631316
// MI455X (gfx1250) — matrix operand order repaired and checked by reading, not compiled, not run on hardware
//
#include <hip/hip_runtime.h>
#include <hip/hip_bf16.h>
#include <math.h>

// ---------------------------------------------------------------------------
// CausalRoPEAttention for MI455X (gfx1250, wave32, WMMA bf16)
//   B=2  S=4096  E=2048  H=16  hd=128  KVH=4  WINDOW=512  N_META=16
// Pipeline (all bf16 WMMA, fp32 accumulate):
//   1. x -> bf16 ; Wq/Wkv/Wo -> transposed bf16 ([N][K])
//   2. Q/KV/O GEMMs: 128x128 block tile, double-buffered LDS staging via async
//      global->LDS copies (ASYNCcnt); RoPE fused into Q/KV epilogues.
//   3. Flash attention in transposed orientation (S^T = K Q^T, O^T = V^T P^T);
//      sched_barriers pin the load clauses so WMMAs only partially wait.
// ---------------------------------------------------------------------------

#define B_SZ     2
#define S_LEN    4096
#define E_DIM    2048
#define HD       128
#define NHEADS   16
#define NKVH     4
#define KV_OUT   1024
#define WINDOW   512
#define NMETA    16

typedef __attribute__((ext_vector_type(16))) __bf16 v16bf;
typedef __attribute__((ext_vector_type(8)))  float  v8f;
typedef __attribute__((ext_vector_type(4)))  unsigned int u32x4;
typedef int v4i __attribute__((__vector_size__(16)));   // matches async builtin param

union Frag16 {
    u32x4          u4[2];
    v16bf          v;
    unsigned short us[16];
};

__device__ __forceinline__ unsigned short f2bf(float f) {
    unsigned int u = __builtin_bit_cast(unsigned int, f);
    unsigned int r = u + 0x7FFFu + ((u >> 16) & 1u);   // round-to-nearest-even
    return (unsigned short)(r >> 16);
}

__device__ __forceinline__ v8f wmma_bf16(const Frag16& a, const Frag16& b, v8f c) {
    return __builtin_amdgcn_wmma_f32_16x16x32_bf16(false, a.v, false, b.v,
                                                   (short)0, c, false, false);
}

// ---------------------- CDNA5 paths & scheduling pins ----------------------

#if defined(__has_builtin)
# if __has_builtin(__builtin_amdgcn_global_load_async_to_lds_b128)
#  define HAVE_ASYNC_LDS 1
# endif
# if __has_builtin(__builtin_amdgcn_s_wait_asynccnt)
#  define HAVE_WAIT_ASYNC 1
# endif
# if __has_builtin(__builtin_amdgcn_sched_barrier)
#  define SCHED_BARRIER() __builtin_amdgcn_sched_barrier(0)
# endif
#endif
#ifndef SCHED_BARRIER
# define SCHED_BARRIER() asm volatile("" ::: "memory")
#endif

__device__ __forceinline__ void copy16_g2l(const unsigned short* g, unsigned short* l) {
#if defined(HAVE_ASYNC_LDS)
    __builtin_amdgcn_global_load_async_to_lds_b128(
        (__attribute__((address_space(1))) v4i*)g,
        (__attribute__((address_space(3))) v4i*)l, 0, 0);
#else
    *(u32x4*)l = *(const u32x4*)g;     // sync fallback: vmem load + ds_store
#endif
}

__device__ __forceinline__ void copy_wait() {
#if defined(HAVE_ASYNC_LDS)
# if defined(HAVE_WAIT_ASYNC)
    __builtin_amdgcn_s_wait_asynccnt(0);
# else
    asm volatile("s_wait_asynccnt 0x0" ::: "memory");
# endif
#endif
}

// --------------------------- conversion kernels ----------------------------

__global__ void k_cvt_bf16(const float* __restrict__ in,
                           unsigned short* __restrict__ out, int n) {
    int i = blockIdx.x * blockDim.x + threadIdx.x;
    int stride = gridDim.x * blockDim.x;
    for (; i < n; i += stride) out[i] = f2bf(in[i]);
}

// in: [K][N] f32 row-major  ->  out: [N][K] bf16 row-major
__global__ void k_transpose_cvt(const float* __restrict__ in,
                                unsigned short* __restrict__ out, int K, int N) {
    __shared__ float tile[32][33];
    const int kb = blockIdx.y * 32, nb = blockIdx.x * 32;
    const int tx = threadIdx.x & 31, ty = threadIdx.x >> 5;   // 8 rows/pass
    for (int r = ty; r < 32; r += 8)
        tile[r][tx] = in[(size_t)(kb + r) * N + (nb + tx)];
    __syncthreads();
    for (int r = ty; r < 32; r += 8)
        out[(size_t)(nb + r) * K + (kb + tx)] = f2bf(tile[tx][r]);
}

// ------------------------------- WMMA GEMM ---------------------------------
// C[M][N] = A[M][K] (bf16) * Bt[N][K]^T (bf16) + bias
// EPI=0: fp32 out   EPI=1: RoPE -> qb   EPI=2: RoPE(K)->kb, V->vtb(transposed)
// block = 256 thr (8 waves); block tile 128(M) x 128(N); k-chunk 32.
// LDS double buffer: sA/sB 2x(128x32) bf16 (32 KB total), filled by async
// global->LDS copies; wave computes 16x128 (8 WMMAs per chunk).

#define ROPE_LN_COEF 0.14391156831f   // ln(10000)/64

template <int EPI>
__global__ __launch_bounds__(256)
void k_gemm_bf16(const unsigned short* __restrict__ A,
                 const unsigned short* __restrict__ Bt,
                 const float* __restrict__ bias,
                 float* __restrict__ Cout,
                 unsigned short* __restrict__ O1,
                 unsigned short* __restrict__ O2,
                 int M, int N, int K) {
    __shared__ unsigned short sA[2][128 * 32];
    __shared__ unsigned short sB[2][128 * 32];

    const int tid  = threadIdx.x;
    const int lane = tid & 31;
    const int wave = tid >> 5;
    const int m0   = blockIdx.y * 128 + wave * 16;
    const int mb   = blockIdx.y * 128;
    const int n0   = blockIdx.x * 128;
    const int nloc = lane & 15;
    const int hi8  = (lane >> 4) << 3;   // 0 | 8
    const int khi  = (lane >> 4) << 4;   // 0 | 16

    v8f acc[8];
    #pragma unroll
    for (int t = 0; t < 8; ++t)
        #pragma unroll
        for (int r = 0; r < 8; ++r) acc[t][r] = 0.f;

    // copy mapping: 512 16B segments each for A and B; 2 of each per thread
    const int r0 = tid >> 2,          s0 = (tid & 3) * 8;          // rows 0..63
    const int r1 = (tid + 256) >> 2,  s1 = s0;                     // rows 64..127

    const int niter = K / 32;

    // prologue: stage k-chunk 0 into buffer 0
    copy16_g2l(A  + (size_t)(mb + r0) * K + s0, &sA[0][r0 * 32 + s0]);
    copy16_g2l(A  + (size_t)(mb + r1) * K + s1, &sA[0][r1 * 32 + s1]);
    copy16_g2l(Bt + (size_t)(n0 + r0) * K + s0, &sB[0][r0 * 32 + s0]);
    copy16_g2l(Bt + (size_t)(n0 + r1) * K + s1, &sB[0][r1 * 32 + s1]);

    for (int kc = 0; kc < niter; ++kc) {
        const int cur = kc & 1;
        copy_wait();
        __syncthreads();                       // tile[cur] visible to all waves
        if (kc + 1 < niter) {                  // stream next tile while computing
            const int k0n = (kc + 1) * 32;
            const int nxt = cur ^ 1;
            copy16_g2l(A  + (size_t)(mb + r0) * K + k0n + s0, &sA[nxt][r0 * 32 + s0]);
            copy16_g2l(A  + (size_t)(mb + r1) * K + k0n + s1, &sA[nxt][r1 * 32 + s1]);
            copy16_g2l(Bt + (size_t)(n0 + r0) * K + k0n + s0, &sB[nxt][r0 * 32 + s0]);
            copy16_g2l(Bt + (size_t)(n0 + r1) * K + k0n + s1, &sB[nxt][r1 * 32 + s1]);
        }
        // compute from tile[cur]: all LDS reads first, then the 8 WMMAs
        Frag16 a;
        const unsigned short* sa = &sA[cur][(wave * 16 + nloc) * 32];
        a.u4[0] = *(const u32x4*)(sa + hi8);
        a.u4[1] = *(const u32x4*)(sa + 16 + hi8);
        Frag16 bfr[8];
        const unsigned short* sb = &sB[cur][nloc * 32 + hi8];
        #pragma unroll
        for (int t = 0; t < 8; ++t) {
            bfr[t].u4[0] = *(const u32x4*)(sb + t * 16 * 32);
            bfr[t].u4[1] = *(const u32x4*)(sb + t * 16 * 32 + 16);
        }
        SCHED_BARRIER();
        #pragma unroll
        for (int t = 0; t < 8; ++t)
            acc[t] = wmma_bf16(a, bfr[t], acc[t]);
    }

    // ------------------------------ epilogue -------------------------------
    #pragma unroll
    for (int t = 0; t < 8; ++t) {
        const int col = n0 + t * 16 + nloc;
        const float bv = bias[col];
        #pragma unroll
        for (int r = 0; r < 8; ++r) {
            const int row = m0 + r + hi8;        // C layout: M = r (+8 hi half)
            float v = acc[t][r] + bv;
            if (EPI == 0) {
                Cout[(size_t)row * N + col] = v;
            } else {
                const float pv = __shfl_xor(v, 1, 32);   // rotate-half partner
                const int s = row & (S_LEN - 1);
                const int bi = row >> 12;                 // row / 4096
                if (EPI == 1) {
                    const int h = col >> 7, d = col & 127, i = d >> 1;
                    float ang = (float)s * __expf(-ROPE_LN_COEF * (float)i);
                    float cs = __cosf(ang), sn = __sinf(ang);
                    float o = (d & 1) ? (v * cs + pv * sn) : (v * cs - pv * sn);
                    O1[(((size_t)(bi * NHEADS + h) * S_LEN + s) * HD) + d] = f2bf(o);
                } else {
                    const int kvsel = col >> 9, w = col & 511;
                    const int kvh = w >> 7, d = w & 127;
                    if (kvsel == 0) {   // K: RoPE
                        const int i = d >> 1;
                        float ang = (float)s * __expf(-ROPE_LN_COEF * (float)i);
                        float cs = __cosf(ang), sn = __sinf(ang);
                        float o = (d & 1) ? (v * cs + pv * sn) : (v * cs - pv * sn);
                        O1[(((size_t)(bi * NKVH + kvh) * S_LEN + s) * HD) + d] = f2bf(o);
                    } else {            // V: transpose to [d][s]
                        O2[(((size_t)(bi * NKVH + kvh) * HD + d) * S_LEN) + s] = f2bf(v);
                    }
                }
            }
        }
    }
}

// ------------------------- flash attention (wave32) ------------------------
// Per wave: 16 queries. Transposed orientation:
//   S^T(16k x 16q) = Kfrag(16k x 32d) * Q^Tfrag(32d x 16q)   [4 WMMA over d]
//   O^T(16d x 16q) += V^Tfrag(16d x 32k) * P^Tfrag(32k x 16q)
// sched_barriers pin the phases: [K loads][score WMMA][V loads][softmax][PV],
// so loads stay clause-grouped and WMMAs use partial loadcnt waits.

__global__ __launch_bounds__(256)
void k_attn(const unsigned short* __restrict__ qb,
            const unsigned short* __restrict__ kbuf,
            const unsigned short* __restrict__ vtb,
            unsigned short* __restrict__ yb) {
    const int lane = threadIdx.x & 31;
    const int wave = threadIdx.x >> 5;
    const int b    = blockIdx.z;
    const int h    = blockIdx.y;
    const int t0   = blockIdx.x * 128 + wave * 16;
    const int kvh  = h >> 2;
    const int nloc = lane & 15;
    const int hi8  = (lane >> 4) << 3;
    const int khi  = (lane >> 4) << 4;
    const int qpos = t0 + nloc;
    const float scale = 0.08838834764831845f;   // 1/sqrt(128)

    const unsigned short* qbase = qb   + (size_t)(b * NHEADS + h) * S_LEN * HD;
    const unsigned short* kbase = kbuf + (size_t)(b * NKVH + kvh) * S_LEN * HD;
    const unsigned short* vbase = vtb  + (size_t)(b * NKVH + kvh) * HD * S_LEN;

    // Preload Q^T B-fragments (lane = query, contraction = d), d0 = 0,32,64,96
    Frag16 qf[4];
    {
        const unsigned short* qrow = qbase + (size_t)(t0 + nloc) * HD;
        #pragma unroll
        for (int t = 0; t < 4; ++t) {
            qf[t].u4[0] = *(const u32x4*)(qrow + t * 32 + hi8);
            qf[t].u4[1] = *(const u32x4*)(qrow + t * 32 + 16 + hi8);
        }
    }

    v8f ot[8];
    #pragma unroll
    for (int t = 0; t < 8; ++t)
        #pragma unroll
        for (int r = 0; r < 8; ++r) ot[t][r] = 0.f;

    float mrun = -1e30f, lrun = 0.f;

    // key blocks: [0,32) (meta+head), then aligned window blocks up to t0+15
    int lo = t0 - (WINDOW - 1); if (lo < 32) lo = 32;
    const int kwstart = lo & ~31;
    const int kend    = t0 + 15;

    int kb0 = 0;
    for (;;) {
        // ----- phase 1: all 16 K loads (one clause, distinct registers) -----
        Frag16 kf[8];     // K-matrix A-fragments: [2 key tiles][4 d-chunks]
        {
            const unsigned short* kr0 = kbase + (size_t)(kb0 + nloc) * HD;
            const unsigned short* kr1 = kr0 + 16 * HD;
            #pragma unroll
            for (int t = 0; t < 4; ++t) {
                kf[t].u4[0]     = *(const u32x4*)(kr0 + t * 32 + hi8);
                kf[t].u4[1]     = *(const u32x4*)(kr0 + t * 32 + 16 + hi8);
                kf[4 + t].u4[0] = *(const u32x4*)(kr1 + t * 32 + hi8);
                kf[4 + t].u4[1] = *(const u32x4*)(kr1 + t * 32 + 16 + hi8);
            }
        }
        SCHED_BARRIER();

        // ----- phase 2: scores S^T (two 16-key tiles) -----
        v8f sc0, sc1;
        #pragma unroll
        for (int r = 0; r < 8; ++r) { sc0[r] = 0.f; sc1[r] = 0.f; }
        #pragma unroll
        for (int t = 0; t < 4; ++t) {
            sc0 = wmma_bf16(kf[t],     qf[t], sc0);
            sc1 = wmma_bf16(kf[4 + t], qf[t], sc1);
        }
        SCHED_BARRIER();

        // ----- phase 3: all 16 V loads (latency hidden under softmax) -----
        Frag16 vf[8];     // V^T A-fragments: one per 16-row d chunk
        #pragma unroll
        for (int t = 0; t < 8; ++t) {
            const unsigned short* vrow = vbase + (size_t)(t * 16 + nloc) * S_LEN;
            vf[t].u4[0] = *(const u32x4*)(vrow + kb0 + hi8);
            vf[t].u4[1] = *(const u32x4*)(vrow + kb0 + 16 + hi8);
        }
        SCHED_BARRIER();

        // ----- phase 4: mask (causal & (window | meta)) + online softmax ----
        float p0[8], p1[8];
        float bm = -1e30f;
        #pragma unroll
        for (int r = 0; r < 8; ++r) {
            const int k0i = kb0 + r + hi8;
            const int k1i = k0i + 16;
            const bool a0 = (k0i <= qpos) && ((k0i + WINDOW - 1 >= qpos) || (k0i < NMETA));
            const bool a1 = (k1i <= qpos) && ((k1i + WINDOW - 1 >= qpos) || (k1i < NMETA));
            p0[r] = a0 ? sc0[r] * scale : -1e30f;
            p1[r] = a1 ? sc1[r] * scale : -1e30f;
            bm = fmaxf(bm, fmaxf(p0[r], p1[r]));
        }
        bm = fmaxf(bm, __shfl_xor(bm, 16, 32));     // other half of key block
        const float mnew  = fmaxf(mrun, bm);
        const float alpha = __expf(mrun - mnew);
        float rsum = 0.f;
        #pragma unroll
        for (int r = 0; r < 8; ++r) {
            p0[r] = __expf(p0[r] - mnew);
            p1[r] = __expf(p1[r] - mnew);
            rsum += p0[r] + p1[r];
        }
        lrun = lrun * alpha + rsum + __shfl_xor(rsum, 16, 32);
        mrun = mnew;
        #pragma unroll
        for (int t = 0; t < 8; ++t)
            #pragma unroll
            for (int r = 0; r < 8; ++r) ot[t][r] *= alpha;

        // build P^T B-fragment directly from this lane's own p0/p1 (no exchange used)
        Frag16 pf;
        #pragma unroll
        for (int e = 0; e < 8; ++e) {
            const float t0p = __shfl_xor(p0[e], 16, 32);
            const float t1p = __shfl_xor(p1[e], 16, 32);
            pf.us[e]     = f2bf(p0[e]);   // keys hi8+0..7
            pf.us[e + 8] = f2bf(p1[e]);   // keys 16+hi8+0..7
        }

        // ----- phase 5: O^T += V^T * P^T -----
        #pragma unroll
        for (int t = 0; t < 8; ++t)
            ot[t] = wmma_bf16(vf[t], pf, ot[t]);

        if (kb0 == 0) kb0 = kwstart; else kb0 += 32;
        if (kb0 > kend) break;
    }

    // ----- normalize, write y (bf16, [b][s][h*hd+d]) -----
    const float linv = 1.0f / lrun;
    unsigned short* ybase =
        yb + ((size_t)b * S_LEN + (t0 + nloc)) * E_DIM + h * HD;
    #pragma unroll
    for (int t = 0; t < 8; ++t) {
        union { unsigned short us[8]; u32x4 u; } pk;
        #pragma unroll
        for (int r = 0; r < 8; ++r) pk.us[r] = f2bf(ot[t][r] * linv);
        *(u32x4*)(ybase + t * 16 + hi8) = pk.u;   // d = t*16 + hi8 + r
    }
}

// -------------------------------- launcher ---------------------------------

extern "C" void kernel_launch(void* const* d_in, const int* in_sizes, int n_in,
                              void* d_out, int out_size, void* d_ws, size_t ws_size,
                              hipStream_t stream) {
    (void)in_sizes; (void)n_in; (void)out_size; (void)ws_size;
    const float* x   = (const float*)d_in[0];
    const float* Wq  = (const float*)d_in[1];
    const float* bq  = (const float*)d_in[2];
    const float* Wkv = (const float*)d_in[3];
    const float* bkv = (const float*)d_in[4];
    const float* Wo  = (const float*)d_in[5];
    const float* bo  = (const float*)d_in[6];
    float* out = (float*)d_out;

    // workspace layout (bytes), ~132 MB total
    char* ws = (char*)d_ws;
    unsigned short* xb   = (unsigned short*)(ws);                 //  33,554,432
    unsigned short* wqt  = (unsigned short*)(ws +  33554432);     //   8,388,608
    unsigned short* wkvt = (unsigned short*)(ws +  41943040);     //   4,194,304
    unsigned short* wot  = (unsigned short*)(ws +  46137344);     //   8,388,608
    unsigned short* qb   = (unsigned short*)(ws +  54525952);     //  33,554,432
    unsigned short* kb   = (unsigned short*)(ws +  88080384);     //   8,388,608
    unsigned short* vtb  = (unsigned short*)(ws +  96468992);     //   8,388,608
    unsigned short* yb   = (unsigned short*)(ws + 104857600);     //  33,554,432

    const int M = B_SZ * S_LEN;                 // 8192
    const int nX = M * E_DIM;                   // 16,777,216

    k_cvt_bf16<<<4096, 256, 0, stream>>>(x, xb, nX);
    k_transpose_cvt<<<dim3(E_DIM / 32,  E_DIM / 32), 256, 0, stream>>>(Wq,  wqt,  E_DIM, E_DIM);
    k_transpose_cvt<<<dim3(KV_OUT / 32, E_DIM / 32), 256, 0, stream>>>(Wkv, wkvt, E_DIM, KV_OUT);
    k_transpose_cvt<<<dim3(E_DIM / 32,  E_DIM / 32), 256, 0, stream>>>(Wo,  wot,  E_DIM, E_DIM);

    k_gemm_bf16<1><<<dim3(E_DIM / 128,  M / 128), 256, 0, stream>>>(
        xb, wqt, bq, nullptr, qb, nullptr, M, E_DIM, E_DIM);
    k_gemm_bf16<2><<<dim3(KV_OUT / 128, M / 128), 256, 0, stream>>>(
        xb, wkvt, bkv, nullptr, kb, vtb, M, KV_OUT, E_DIM);

    k_attn<<<dim3(S_LEN / 128, NHEADS, B_SZ), 256, 0, stream>>>(qb, kb, vtb, yb);

    k_gemm_bf16<0><<<dim3(E_DIM / 128, M / 128), 256, 0, stream>>>(
        yb, wot, bo, out, nullptr, nullptr, M, E_DIM, E_DIM);
}
